// SSEGCNClassifier_7284264534271
// MI455X (gfx1250) — matrix operand order repaired and checked by reading, not compiled, not run on hardware
//
#include <hip/hip_runtime.h>
#include <math.h>

#define DEV __device__ __forceinline__

namespace ssegcn {

constexpr int B   = 64;
constexpr int N   = 128;
constexpr int H   = 6;
constexpr int ED  = 300;
constexpr int PD  = 30;
constexpr int RH  = 300;           // rnn hidden
constexpr int D   = 2 * RH;        // 600
constexpr int DK  = D / H;         // 100
constexpr int IN_DIM = ED + 2*PD;  // 360
constexpr int G4  = 4 * RH;        // 1200 (gate width)
constexpr int POL = 3;
constexpr float RSQRT_DK = 0.1f;   // 1/sqrt(100)

typedef __attribute__((ext_vector_type(16))) _Float16 v16h;
typedef __attribute__((ext_vector_type(8)))  float    v8f;

// ---------------------------------------------------------------------------
// WMMA fragment loaders (f32 memory -> f16 fragments), all BRANCH-FREE.
// Row / column indices are clamped (out-of-range rows/cols only pollute C
// entries that are never stored).  K-bounds are handled only in the tail
// fragment via value selects (cndmask), never via divergent branches.
// ---------------------------------------------------------------------------

// A operand 16x32 (MxK) of row-major p[Mr x K]; no K masking (full chunk).
DEV v16h frag_a_fast(const float* p, int ld, int rbase, int kb, int Mr) {
  int l = threadIdx.x & 31;
  int r = rbase + (l & 15);
  r = (r < Mr) ? r : (Mr - 1);                 // clamp, no branch
  const float* row = p + (size_t)r * ld + kb + ((l >> 4) << 4);
  v16h f;
#pragma unroll
  for (int e = 0; e < 16; ++e) f[e] = (_Float16)row[e];          // k: +0..15
  return f;
}

// A operand tail chunk with K masking.
DEV v16h frag_a_tail(const float* p, int ld, int rbase, int kb, int Mr, int K) {
  int l = threadIdx.x & 31;
  int r = rbase + (l & 15);
  r = (r < Mr) ? r : (Mr - 1);
  const float* row = p + (size_t)r * ld;
  int kofs = kb + ((l >> 4) << 4);
  v16h f;
#pragma unroll
  for (int e = 0; e < 16; ++e) {
    int k  = kofs + e;
    int kc = (k < K) ? k : 0;                  // clamp address
    float v = row[kc];                         // unconditional load
    f[e] = (_Float16)((k < K) ? v : 0.0f);     // mask value
  }
  return f;
}

// B operand (KxN) from W row-major [Nc x K]  (C = A @ W^T); full chunk.
DEV v16h frag_bt_fast(const float* p, int ld, int nbase, int kb, int Nc) {
  int l = threadIdx.x & 31;
  int n = nbase + (l & 15);
  n = (n < Nc) ? n : (Nc - 1);
  const float* row = p + (size_t)n * ld + kb + ((l >> 4) << 4);
  v16h f;
#pragma unroll
  for (int e = 0; e < 16; ++e) f[e] = (_Float16)row[e];          // k: +0..15
  return f;
}

DEV v16h frag_bt_tail(const float* p, int ld, int nbase, int kb, int Nc, int K) {
  int l = threadIdx.x & 31;
  int n = nbase + (l & 15);
  n = (n < Nc) ? n : (Nc - 1);
  const float* row = p + (size_t)n * ld;
  int kofs = kb + ((l >> 4) << 4);
  v16h f;
#pragma unroll
  for (int e = 0; e < 16; ++e) {
    int k  = kofs + e;
    int kc = (k < K) ? k : 0;
    float v = row[kc];
    f[e] = (_Float16)((k < K) ? v : 0.0f);
  }
  return f;
}

// B operand from row-major [K x Nc]  (C = A @ Bm); full chunk.
DEV v16h frag_bn_fast(const float* p, int ld, int nbase, int kb, int Nc) {
  int l = threadIdx.x & 31;
  int n = nbase + (l & 15);
  n = (n < Nc) ? n : (Nc - 1);
  const float* col = p + n + (size_t)(kb + ((l >> 4) << 4)) * ld;
  v16h f;
#pragma unroll
  for (int e = 0; e < 16; ++e) f[e] = (_Float16)col[(size_t)e * ld];
  return f;
}

DEV v16h frag_bn_tail(const float* p, int ld, int nbase, int kb, int Nc, int K) {
  int l = threadIdx.x & 31;
  int n = nbase + (l & 15);
  n = (n < Nc) ? n : (Nc - 1);
  int kofs = kb + ((l >> 4) << 4);
  v16h f;
#pragma unroll
  for (int e = 0; e < 16; ++e) {
    int k  = kofs + e;
    int kc = (k < K) ? k : 0;
    float v = p[(size_t)kc * ld + n];
    f[e] = (_Float16)((k < K) ? v : 0.0f);
  }
  return f;
}

// ---------------------------------------------------------------------------
// Generic batched WMMA GEMM.  One wave computes a 16x32 C tile (one A
// fragment feeds two accumulators).  K-loop: mask-free main body + one
// masked tail chunk.
// C[z] = act( alpha * A[z] (.) B[z] + bias )
// TRANSB=1: B operand is W[Nc x K] row-major (A @ W^T). TRANSB=0: B is [K x Nc].
// ACT: 0 = none, 1 = SELU, 2 = ReLU.
// ---------------------------------------------------------------------------
template<int TRANSB, int ACT>
__global__ void __launch_bounds__(32)
gemm_wmma(const float* __restrict__ A, int lda, long long sA,
          const float* __restrict__ Bm, int ldb, long long sB,
          float* __restrict__ C, int ldc, long long sC,
          const float* __restrict__ bias,
          int M, int Nc, int K, float alpha)
{
  const float* Ab = A  + (size_t)blockIdx.z * sA;
  const float* Bb = Bm + (size_t)blockIdx.z * sB;
  float*       Cb = C  + (size_t)blockIdx.z * sC;
  int tm  = blockIdx.y << 4;
  int tn0 = blockIdx.x << 5;
  int tn1 = tn0 + 16;
  v8f acc0 = {};
  v8f acc1 = {};
  const int Kmain = K & ~31;
#pragma unroll 2
  for (int kb = 0; kb < Kmain; kb += 32) {
    v16h a  = frag_a_fast(Ab, lda, tm, kb, M);
    v16h b0 = TRANSB ? frag_bt_fast(Bb, ldb, tn0, kb, Nc)
                     : frag_bn_fast(Bb, ldb, tn0, kb, Nc);
    v16h b1 = TRANSB ? frag_bt_fast(Bb, ldb, tn1, kb, Nc)
                     : frag_bn_fast(Bb, ldb, tn1, kb, Nc);
    acc0 = __builtin_amdgcn_wmma_f32_16x16x32_f16(false, a, false, b0,
                                                  (short)0, acc0, false, false);
    acc1 = __builtin_amdgcn_wmma_f32_16x16x32_f16(false, a, false, b1,
                                                  (short)0, acc1, false, false);
  }
  if (Kmain < K) {
    v16h a  = frag_a_tail(Ab, lda, tm, Kmain, M, K);
    v16h b0 = TRANSB ? frag_bt_tail(Bb, ldb, tn0, Kmain, Nc, K)
                     : frag_bn_tail(Bb, ldb, tn0, Kmain, Nc, K);
    v16h b1 = TRANSB ? frag_bt_tail(Bb, ldb, tn1, Kmain, Nc, K)
                     : frag_bn_tail(Bb, ldb, tn1, Kmain, Nc, K);
    acc0 = __builtin_amdgcn_wmma_f32_16x16x32_f16(false, a, false, b0,
                                                  (short)0, acc0, false, false);
    acc1 = __builtin_amdgcn_wmma_f32_16x16x32_f16(false, a, false, b1,
                                                  (short)0, acc1, false, false);
  }
  int l     = threadIdx.x & 31;
  int lane  = l & 15;
  int rbase = tm + ((l >> 4) << 3);
  int col0  = tn0 + lane;
  int col1  = tn1 + lane;
  float bv0 = 0.0f, bv1 = 0.0f;
  if (bias != nullptr) {
    bv0 = bias[(col0 < Nc) ? col0 : (Nc - 1)];
    bv1 = bias[(col1 < Nc) ? col1 : (Nc - 1)];
  }
#pragma unroll
  for (int r = 0; r < 8; ++r) {
    int row = rbase + r;
    float v0 = fmaf(acc0[r], alpha, bv0);
    float v1 = fmaf(acc1[r], alpha, bv1);
    if (ACT == 1) {            // SELU
      const float sc = 1.0507009873554805f, al = 1.6732632423543772f;
      v0 = (v0 > 0.0f) ? sc * v0 : sc * al * (__expf(v0) - 1.0f);
      v1 = (v1 > 0.0f) ? sc * v1 : sc * al * (__expf(v1) - 1.0f);
    } else if (ACT == 2) {     // ReLU
      v0 = fmaxf(v0, 0.0f);
      v1 = fmaxf(v1, 0.0f);
    }
    if (row < M) {
      if (col0 < Nc) Cb[(size_t)row * ldc + col0] = v0;
      if (col1 < Nc) Cb[(size_t)row * ldc + col1] = v1;
    }
  }
}

// ---- Embedding concat ------------------------------------------------------
__global__ void embed_kernel(const int* __restrict__ tok,
                             const int* __restrict__ pos_ids,
                             const int* __restrict__ post_ids,
                             const float* __restrict__ emb_w,
                             const float* __restrict__ pos_w,
                             const float* __restrict__ post_w,
                             float* __restrict__ x)
{
  int i = blockIdx.x * blockDim.x + threadIdx.x;
  if (i >= B * N * IN_DIM) return;
  int c  = i % IN_DIM;
  int bn = i / IN_DIM;
  float v;
  if (c < ED)           v = emb_w [(size_t)tok[bn]      * ED + c];
  else if (c < ED + PD) v = pos_w [(size_t)pos_ids[bn]  * PD + (c - ED)];
  else                  v = post_w[(size_t)post_ids[bn] * PD + (c - ED - PD)];
  x[i] = v;
}

// ---- LSTM ------------------------------------------------------------------
DEV float sigmoidf_(float v) { return 1.0f / (1.0f + __expf(-v)); }

// One recurrence step for both directions. grid.x = 2*B (dir,b), block = 320.
__global__ void __launch_bounds__(320)
lstm_step_kernel(const float* __restrict__ xg_f, const float* __restrict__ xg_b,
                 const float* __restrict__ Whh_f, const float* __restrict__ Whh_b,
                 float* __restrict__ g_in, float* __restrict__ hstate,
                 float* __restrict__ cstate, int t)
{
  __shared__ float hs[RH];
  int dir = blockIdx.x >> 6;           // B == 64
  int b   = blockIdx.x & 63;
  int r   = threadIdx.x;
  float* hrow = hstate + ((size_t)dir * B + b) * RH;
  float* crow = cstate + ((size_t)dir * B + b) * RH;
  if (r < RH) hs[r] = hrow[r];
  __syncthreads();
  if (r >= RH) return;
  int trow = dir ? (N - 1 - t) : t;    // bwd dir walks the sequence reversed
  const float* xg  = (dir ? xg_b : xg_f) + ((size_t)b * N + trow) * G4;
  const float* Whh = dir ? Whh_b : Whh_f;
  float gi = xg[r], gf = xg[RH + r], gc = xg[2*RH + r], go = xg[3*RH + r];
  const float* wi = Whh + (size_t)(r)        * RH;
  const float* wf = Whh + (size_t)(RH + r)   * RH;
  const float* wc = Whh + (size_t)(2*RH + r) * RH;
  const float* wo = Whh + (size_t)(3*RH + r) * RH;
  for (int k = 0; k < RH; ++k) {
    float h = hs[k];
    gi = fmaf(wi[k], h, gi);
    gf = fmaf(wf[k], h, gf);
    gc = fmaf(wc[k], h, gc);
    go = fmaf(wo[k], h, go);
  }
  float c = sigmoidf_(gf) * crow[r] + sigmoidf_(gi) * tanhf(gc);
  float h = sigmoidf_(go) * tanhf(c);
  crow[r] = c;
  hrow[r] = h;
  // write directly into the concat layout g_in[b, trow, dir*RH + r]
  g_in[((size_t)b * N + trow) * D + dir * RH + r] = h;
}

// ---- masked mean over N:  out[b,d] = sum_n src[b,n,d]*mask[b,n] / sum_n mask
__global__ void masked_mean_kernel(const float* __restrict__ src,
                                   const float* __restrict__ mask,
                                   float* __restrict__ out, int Dd)
{
  int i = blockIdx.x * blockDim.x + threadIdx.x;
  if (i >= B * Dd) return;
  int b = i / Dd, d = i % Dd;
  float s = 0.0f, wn = 0.0f;
  for (int n = 0; n < N; ++n) {
    float m = mask[b * N + n];
    s  += src[((size_t)b * N + n) * Dd + d] * m;
    wn += m;
  }
  out[i] = s / wn;
}

// ---- repack [B,N,H*DK] -> [B,H,N,DK] --------------------------------------
__global__ void repack_heads(const float* __restrict__ in, float* __restrict__ out)
{
  int i = blockIdx.x * blockDim.x + threadIdx.x;
  if (i >= B * N * D) return;
  int c = i % D;
  int n = (i / D) % N;
  int b = i / (N * D);
  int h = c / DK, e = c % DK;
  out[(((size_t)b * H + h) * N + n) * DK + e] = in[i];
}

// ---- asc[b,h,m] = tanh( aw[b,h,:] . Kh[b,h,m,:] + bias_m ) -----------------
__global__ void asc_kernel(const float* __restrict__ aw, const float* __restrict__ Kh,
                           const float* __restrict__ bias_m, float* __restrict__ ascv)
{
  int i = blockIdx.x * blockDim.x + threadIdx.x;
  if (i >= B * H * N) return;
  int m  = i % N;
  int bh = i / N;
  const float* a = aw + (size_t)bh * DK;
  const float* k = Kh + ((size_t)bh * N + m) * DK;
  float s = bias_m[0];
  for (int e = 0; e < DK; ++e) s = fmaf(a[e], k[e], s);
  ascv[i] = tanhf(s);
}

// ---- scores assembly + softmax: one block per (b,h,n) row ------------------
__global__ void __launch_bounds__(N)
attn_softmax_kernel(const float* __restrict__ raw, const float* __restrict__ ascv,
                    const int* __restrict__ tok, const float* __restrict__ shortb,
                    float* __restrict__ adj)
{
  __shared__ float red[N];
  int n  = blockIdx.x % N;
  int bh = blockIdx.x / N;
  int b  = bh / H;
  int m  = threadIdx.x;
  size_t off = ((size_t)bh * N + n) * N + m;
  float s = raw[off] * RSQRT_DK + ascv[(size_t)bh * N + m];
  if (tok[b * N + m] == 0) s = -1e9f;
  s += shortb[((size_t)b * N + n) * N + m];     // short is [B,1,N,N]
  red[m] = s;
  __syncthreads();
  for (int st = N / 2; st > 0; st >>= 1) {
    if (m < st) red[m] = fmaxf(red[m], red[m + st]);
    __syncthreads();
  }
  float e = __expf(s - red[0]);
  __syncthreads();
  red[m] = e;
  __syncthreads();
  for (int st = N / 2; st > 0; st >>= 1) {
    if (m < st) red[m] += red[m + st];
    __syncthreads();
  }
  adj[off] = e / red[0];
}

// ---- adjacency sum over heads ---------------------------------------------
__global__ void adjsum_kernel(const float* __restrict__ adj, float* __restrict__ out)
{
  int i = blockIdx.x * blockDim.x + threadIdx.x;
  if (i >= B * N * N) return;
  int b  = i / (N * N);
  int nm = i % (N * N);
  float s = 0.0f;
  for (int h = 0; h < H; ++h) s += adj[((size_t)b * H + h) * (N * N) + nm];
  out[i] = s;
}

// ---- fused edge update -----------------------------------------------------
// out[b,k,n,m] = sum_h adj[b,h,n,m]*Wx[k,h] + gn1[b,m,k] + gn2[b,n,k] + bx[k]
__global__ void edge_update_kernel(const float* __restrict__ adj,
                                   const float* __restrict__ gn1,
                                   const float* __restrict__ gn2,
                                   const float* __restrict__ Wx,
                                   const float* __restrict__ bx,
                                   float* __restrict__ out)
{
  int i = blockIdx.x * blockDim.x + threadIdx.x;
  if (i >= B * N * N) return;
  int b  = i / (N * N);
  int nm = i % (N * N);
  int n  = nm / N, m = nm % N;
  float a[H];
#pragma unroll
  for (int h = 0; h < H; ++h) a[h] = adj[((size_t)b * H + h) * (N * N) + nm];
  const float* g1p = gn1 + ((size_t)b * N + m) * H;
  const float* g2p = gn2 + ((size_t)b * N + n) * H;
#pragma unroll
  for (int k = 0; k < H; ++k) {
    float s = bx[k] + g1p[k] + g2p[k];
#pragma unroll
    for (int h = 0; h < H; ++h) s = fmaf(a[h], Wx[(size_t)k * (H + 2 * D) + h], s);
    out[((size_t)b * H + k) * (N * N) + nm] = s;
  }
}

} // namespace ssegcn

using namespace ssegcn;

static inline int cdiv(int a, int b) { return (a + b - 1) / b; }

extern "C" void kernel_launch(void* const* d_in, const int* in_sizes, int n_in,
                              void* d_out, int out_size, void* d_ws, size_t ws_size,
                              hipStream_t stream)
{
  (void)in_sizes; (void)n_in; (void)out_size; (void)ws_size;

  const int*   tok      = (const int*)  d_in[0];
  const int*   pos_ids  = (const int*)  d_in[2];
  const int*   post_ids = (const int*)  d_in[5];
  const float* mask     = (const float*)d_in[6];
  const float* short_   = (const float*)d_in[8];
  const float* emb_w    = (const float*)d_in[9];
  const float* pos_w    = (const float*)d_in[10];
  const float* post_w   = (const float*)d_in[11];
  const float* Wih_f    = (const float*)d_in[12];
  const float* Whh_f    = (const float*)d_in[13];
  const float* b_f      = (const float*)d_in[14];
  const float* Wih_b    = (const float*)d_in[15];
  const float* Whh_b    = (const float*)d_in[16];
  const float* b_b      = (const float*)d_in[17];
  const float* Wq       = (const float*)d_in[18];
  const float* bq       = (const float*)d_in[19];
  const float* Wk       = (const float*)d_in[20];
  const float* bk       = (const float*)d_in[21];
  const float* Wd       = (const float*)d_in[22];
  const float* bd       = (const float*)d_in[23];
  const float* weight_m = (const float*)d_in[24];
  const float* bias_m   = (const float*)d_in[25];
  const float* Ww       = (const float*)d_in[26];
  const float* bw       = (const float*)d_in[27];
  const float* Wx       = (const float*)d_in[28];
  const float* bx       = (const float*)d_in[29];
  const float* Wxx      = (const float*)d_in[30];
  const float* bxx      = (const float*)d_in[31];
  const float* Wc       = (const float*)d_in[32];
  const float* bc       = (const float*)d_in[33];
  float* out = (float*)d_out;

  // ---- workspace layout (fp32 elements) ----
  float* ws = (float*)d_ws;
  size_t off = 0;
  auto alloc = [&](size_t nelem) { float* p = ws + off; off += nelem; return p; };
  float* x      = alloc((size_t)B * N * IN_DIM);
  float* xg_f   = alloc((size_t)B * N * G4);
  float* xg_b   = alloc((size_t)B * N * G4);
  float* g_in   = alloc((size_t)B * N * D);
  float* st_h   = alloc((size_t)2 * B * RH);
  float* st_c   = alloc((size_t)2 * B * RH);   // contiguous after st_h
  float* aspect = alloc((size_t)B * D);
  float* rawsc  = alloc((size_t)B * H * N * N);
  float* adj    = alloc((size_t)B * H * N * N);
  float* a_v    = alloc((size_t)B * DK);
  float* aw     = alloc((size_t)B * H * DK);
  float* ascv   = alloc((size_t)B * H * N);
  float* adjsum = alloc((size_t)B * N * N);
  float* Axb    = alloc((size_t)B * N * D);
  float* g1     = alloc((size_t)B * N * D);
  float* g2     = alloc((size_t)B * N * D);
  float* gn1    = alloc((size_t)B * N * H);
  float* gn2    = alloc((size_t)B * N * H);
  float* hnode  = alloc((size_t)B * N * RH);
  float* pooled = alloc((size_t)B * RH);
  // Exact-fit aliases of dead regions:
  float* Qm   = xg_f;                           // xg dead after recurrence
  float* Km   = xg_f + (size_t)B * N * D;       // 2*(B*N*D) == B*N*G4
  float* Qh   = xg_b;
  float* Kh   = xg_b + (size_t)B * N * D;
  float* adj2 = rawsc;                          // raw scores dead after softmax

  // 1. embedding concat
  {
    int tot = B * N * IN_DIM;
    embed_kernel<<<cdiv(tot, 256), 256, 0, stream>>>(tok, pos_ids, post_ids,
                                                     emb_w, pos_w, post_w, x);
  }
  // 2. LSTM input gates: xg = x @ Wih^T + b   [B*N, 1200]
  {
    dim3 g(cdiv(G4, 32), cdiv(B * N, 16), 1);
    gemm_wmma<1, 0><<<g, 32, 0, stream>>>(x, IN_DIM, 0, Wih_f, IN_DIM, 0,
                                          xg_f, G4, 0, b_f, B * N, G4, IN_DIM, 1.0f);
    gemm_wmma<1, 0><<<g, 32, 0, stream>>>(x, IN_DIM, 0, Wih_b, IN_DIM, 0,
                                          xg_b, G4, 0, b_b, B * N, G4, IN_DIM, 1.0f);
  }
  // 3. recurrence (both directions fused per step); zero h0/c0 first
  hipMemsetAsync(st_h, 0, (size_t)4 * B * RH * sizeof(float), stream);
  for (int t = 0; t < N; ++t)
    lstm_step_kernel<<<2 * B, 320, 0, stream>>>(xg_f, xg_b, Whh_f, Whh_b,
                                                g_in, st_h, st_c, t);
  // 4. aspect pooling
  masked_mean_kernel<<<cdiv(B * D, 256), 256, 0, stream>>>(g_in, mask, aspect, D);
  // 5. Q, K projections (+ repack to [B,H,N,DK])
  {
    dim3 g(cdiv(D, 32), cdiv(B * N, 16), 1);
    gemm_wmma<1, 0><<<g, 32, 0, stream>>>(g_in, D, 0, Wq, D, 0, Qm, D, 0, bq,
                                          B * N, D, D, 1.0f);
    gemm_wmma<1, 0><<<g, 32, 0, stream>>>(g_in, D, 0, Wk, D, 0, Km, D, 0, bk,
                                          B * N, D, D, 1.0f);
  }
  repack_heads<<<cdiv(B * N * D, 256), 256, 0, stream>>>(Qm, Qh);
  repack_heads<<<cdiv(B * N * D, 256), 256, 0, stream>>>(Km, Kh);
  // 6. raw attention scores: per (b,h) 128x128 = Qh @ Kh^T
  {
    dim3 g(cdiv(N, 32), cdiv(N, 16), B * H);
    gemm_wmma<1, 0><<<g, 32, 0, stream>>>(Qh, DK, (long long)N * DK,
                                          Kh, DK, (long long)N * DK,
                                          rawsc, N, (long long)N * N,
                                          nullptr, N, N, DK, 1.0f);
  }
  // 7. a = aspect @ Wd^T + bd   [B, DK]
  {
    dim3 g(cdiv(DK, 32), cdiv(B, 16), 1);
    gemm_wmma<1, 0><<<g, 32, 0, stream>>>(aspect, D, 0, Wd, D, 0, a_v, DK, 0, bd,
                                          B, DK, D, 1.0f);
  }
  // 8. aw[b,h,:] = a[b] @ weight_m[h]  (batched over h, notrans B)
  {
    dim3 g(cdiv(DK, 32), cdiv(B, 16), H);
    gemm_wmma<0, 0><<<g, 32, 0, stream>>>(a_v, DK, 0,
                                          weight_m, DK, (long long)DK * DK,
                                          aw, H * DK, (long long)DK,
                                          nullptr, B, DK, DK, 1.0f);
  }
  // 9. asc = tanh(aw . K + bias_m)
  asc_kernel<<<cdiv(B * H * N, 256), 256, 0, stream>>>(aw, Kh, bias_m, ascv);
  // 10. scores assembly + softmax -> adj
  attn_softmax_kernel<<<B * H * N, N, 0, stream>>>(rawsc, ascv, tok, short_, adj);
  // 11. GCN layer 1
  adjsum_kernel<<<cdiv(B * N * N, 256), 256, 0, stream>>>(adj, adjsum);
  {
    dim3 g(cdiv(D, 32), cdiv(N, 16), B);
    gemm_wmma<0, 0><<<g, 32, 0, stream>>>(adjsum, N, (long long)N * N,
                                          g_in, D, (long long)N * D,
                                          Axb, D, (long long)N * D,
                                          nullptr, N, D, N, 1.0f / H);
  }
  {
    dim3 g(cdiv(D, 32), cdiv(B * N, 16), 1);
    gemm_wmma<1, 1><<<g, 32, 0, stream>>>(Axb, D, 0, Ww, D, 0, g1, D, 0, bw,
                                          B * N, D, D, 1.0f);   // SELU
  }
  {
    dim3 g(cdiv(H, 32), cdiv(B * N, 16), 1);
    gemm_wmma<1, 0><<<g, 32, 0, stream>>>(g1, D, 0, Wx + H, H + 2 * D, 0,
                                          gn1, H, 0, nullptr, B * N, H, D, 1.0f);
    gemm_wmma<1, 0><<<g, 32, 0, stream>>>(g1, D, 0, Wx + H + D, H + 2 * D, 0,
                                          gn2, H, 0, nullptr, B * N, H, D, 1.0f);
  }
  edge_update_kernel<<<cdiv(B * N * N, 256), 256, 0, stream>>>(adj, gn1, gn2,
                                                               Wx, bx, adj2);
  // 12. GCN layer 2 (edge update of layer 2 is discarded by the reference)
  adjsum_kernel<<<cdiv(B * N * N, 256), 256, 0, stream>>>(adj2, adjsum);
  {
    dim3 g(cdiv(D, 32), cdiv(N, 16), B);
    gemm_wmma<0, 0><<<g, 32, 0, stream>>>(adjsum, N, (long long)N * N,
                                          g1, D, (long long)N * D,
                                          Axb, D, (long long)N * D,
                                          nullptr, N, D, N, 1.0f / H);
  }
  {
    dim3 g(cdiv(D, 32), cdiv(B * N, 16), 1);
    gemm_wmma<1, 1><<<g, 32, 0, stream>>>(Axb, D, 0, Ww, D, 0, g2, D, 0, bw,
                                          B * N, D, D, 1.0f);   // SELU
  }
  // 13. hnode = relu(g2 @ Wxx^T + bxx)
  {
    dim3 g(cdiv(RH, 32), cdiv(B * N, 16), 1);
    gemm_wmma<1, 2><<<g, 32, 0, stream>>>(g2, D, 0, Wxx, D, 0, hnode, RH, 0, bxx,
                                          B * N, RH, D, 1.0f); // ReLU
  }
  // 14. pooled
  masked_mean_kernel<<<cdiv(B * RH, 256), 256, 0, stream>>>(hnode, mask, pooled, RH);
  // 15. logits = pooled @ Wc^T + bc  -> d_out [B, 3]
  {
    dim3 g(cdiv(POL, 32), cdiv(B, 16), 1);
    gemm_wmma<1, 0><<<g, 32, 0, stream>>>(pooled, RH, 0, Wc, RH, 0, out, POL, 0, bc,
                                          B, POL, RH, 1.0f);
  }
}
